// Attention_58171037057295
// MI455X (gfx1250) — matrix operand order repaired and checked by reading, not compiled, not run on hardware
//
#include <hip/hip_runtime.h>

// ---------------------------------------------------------------------------
// MI455X (gfx1250) attention, all matmuls on V_WMMA_F32_16X16X32_BF16.
//   cvt(fp32->bf16) -> GEMM(Q +RoPE +0.125) / GEMM(K +RoPE) / GEMM(V^T)
//   -> flash attention (QK^T, online softmax, PV) -> GEMM(O, fp32 out)
// Global->LDS staging uses CDNA5 async direct-to-LDS (ASYNCcnt) with LDS
// ping-pong so the next tile's DMA overlaps the current tile's WMMA work.
// ---------------------------------------------------------------------------

typedef __attribute__((ext_vector_type(16))) __bf16         v16bf;
typedef __attribute__((ext_vector_type(8)))  float          v8f;
typedef __attribute__((ext_vector_type(4)))  float          f32x4;
typedef __attribute__((ext_vector_type(4)))  unsigned short u16x4;

#define S_LEN 2048
#define H_DIM 2048
#define HD    64
#define NH    32
#define KVH   8

__device__ __forceinline__ unsigned short f2bf(float f) {
  unsigned int u = __float_as_uint(f);
  u += 0x7fffu + ((u >> 16) & 1u);        // round-to-nearest-even
  return (unsigned short)(u >> 16);
}

union FragU { v16bf v; f32x4 q[2]; };

// A-fragment (16x32 bf16), ISA 7.12.2: M = lane&15;
// lanes 0-15: V0-3=K0..7, V4-7=K16..23 ; lanes 16-31: V0-3=K8..15, V4-7=K24..31
__device__ __forceinline__ v16bf load_frag_a(const unsigned short* p0, int stride,
                                             int k0, int lane) {
  const int lc = lane & 15, hf = lane >> 4;
  const unsigned short* p = p0 + lc * stride + k0 + hf * 8;
  FragU u;
  u.q[0] = *(const f32x4*)(p);
  u.q[1] = *(const f32x4*)(p + 16);
  return u.v;
}

// B-fragment (32x16 bf16) from a [N][K]-layout tile: N = lane&15;
// lanes 0-15 hold K 0..7,16..23 ; lanes 16-31 hold K 8..15,24..31 (same as A).
__device__ __forceinline__ v16bf load_frag_b(const unsigned short* p0, int colBase,
                                             int stride, int k0, int lane) {
  const int lc = lane & 15, hf = lane >> 4;
  const unsigned short* p = p0 + (colBase + lc) * stride + k0 + hf * 8;
  FragU u;
  u.q[0] = *(const f32x4*)(p);
  u.q[1] = *(const f32x4*)(p + 16);
  return u.v;
}

__device__ __forceinline__ v8f wmma_bf16(v16bf a, v16bf b, v8f c) {
  return __builtin_amdgcn_wmma_f32_16x16x32_bf16(false, a, false, b, (short)0, c,
                                                 false, false);
}

// CDNA5 async global->LDS copy (GVS mode): per-lane 16B deposited straight to
// LDS, tracked by ASYNCcnt. lds_off/gbyte_off are byte offsets; sbase is the
// wave-uniform 64-bit base (SGPR pair).
__device__ __forceinline__ void async_ld_b128(unsigned lds_off, const void* sbase,
                                              unsigned gbyte_off) {
  asm volatile("global_load_async_to_lds_b128 %0, %1, %2"
               :: "v"(lds_off), "v"(gbyte_off), "s"(sbase) : "memory");
}
__device__ __forceinline__ void wait_async() {
  asm volatile("s_wait_asynccnt 0x0" ::: "memory");
}

// ---------------------------------------------------------------------------
// fp32 -> bf16 conversion (x4 vectorized)
// ---------------------------------------------------------------------------
__global__ void cvt_bf16_v4(const float* __restrict__ in,
                            unsigned short* __restrict__ out, int n4) {
  int i = blockIdx.x * blockDim.x + threadIdx.x;
  if (i < n4) {
    f32x4 x = ((const f32x4*)in)[i];
    u16x4 r;
    r.x = f2bf(x.x); r.y = f2bf(x.y); r.z = f2bf(x.z); r.w = f2bf(x.w);
    ((u16x4*)out)[i] = r;
  }
}

// ---------------------------------------------------------------------------
// WMMA GEMM: C[2048, N] = A[2048, K] * B[K, N], bf16 in / fp32 accum.
// Workgroup tile 128x64, 4 waves; wave tile 32x64 => 8 WMMA / 12 ds_load_b128
// per 32-K chunk. Double-buffered LDS: next A-tile lands via async direct-to-
// LDS (and next B-tile is transposed through VGPRs) while current chunk's
// WMMAs execute; ASYNCcnt drained only after compute.
// MODE 0: Q  -> RoPE + 0.125 folded, bf16 [S][H]
// MODE 1: K  -> RoPE, bf16 [KVH][S][HD]
// MODE 2: V  -> bf16 transposed [KVH][HD][S]
// MODE 3: O  -> fp32 [S][H] (final output)
// ---------------------------------------------------------------------------
template <int MODE>
__global__ __launch_bounds__(128) void gemm_wmma(
    const unsigned short* __restrict__ A, const unsigned short* __restrict__ B,
    unsigned short* __restrict__ outb, float* __restrict__ outf, int N, int K) {
  // per buffer: A [128][32] (8KB) + B [64][32] (4KB); two buffers = 24KB
  __shared__ unsigned short smem[2 * 6144];
  const int tid = threadIdx.x, lane = tid & 31, w = tid >> 5;
  const int rowBase = blockIdx.y * 128, colBase = blockIdx.x * 64;

  v8f acc[2][4];
#pragma unroll
  for (int t = 0; t < 2; ++t)
#pragma unroll
    for (int c = 0; c < 4; ++c) acc[t][c] = (v8f)(0.0f);

  auto stage = [&](int k0, int buf) {
    // A tile 128x32: one row per thread, 4 async b128 straight into LDS
    const unsigned lbase = (unsigned)(buf * 12288);  // LDS byte base of buffer
    const unsigned gb = (unsigned)(((size_t)(rowBase + tid) * K + k0) * 2);
#pragma unroll
    for (int j = 0; j < 4; ++j)
      async_ld_b128(lbase + (unsigned)(tid * 64 + j * 16), A, gb + j * 16);
    // B tile transposed into [n][k] while async A is in flight
    const int kk = tid >> 2, nb = (tid & 3) * 16;
    const unsigned short* src = B + (size_t)(k0 + kk) * N + colBase + nb;
    if (k0 + 32 < K) __builtin_prefetch(src + 32 * N);
    unsigned short* bs = smem + buf * 6144 + 4096;
#pragma unroll
    for (int j = 0; j < 16; ++j) bs[(nb + j) * 32 + kk] = src[j];
  };

  stage(0, 0);
  wait_async();
  __syncthreads();

  const int nk = K / 32;
  for (int ki = 0; ki < nk; ++ki) {
    const int cur = ki & 1;
    if (ki + 1 < nk) stage((ki + 1) * 32, cur ^ 1);  // overlap with compute

    const unsigned short* As = smem + cur * 6144;
    const unsigned short* Bs = As + 4096;
    v16bf a0 = load_frag_a(&As[(w * 32) * 32], 32, 0, lane);
    v16bf a1 = load_frag_a(&As[(w * 32 + 16) * 32], 32, 0, lane);
#pragma unroll
    for (int c = 0; c < 4; ++c) {
      v16bf b = load_frag_b(Bs, c * 16, 32, 0, lane);
      acc[0][c] = wmma_bf16(a0, b, acc[0][c]);
      acc[1][c] = wmma_bf16(a1, b, acc[1][c]);
    }
    wait_async();     // next buffer's DMA done (was in flight during compute)
    __syncthreads();  // all waves done reading cur, deposits visible
  }

  // Epilogue. C/D layout: row = 8*(lane>=16)+v, col = lane&15 (+16*c).
  const int lc = lane & 15;
  const int rh = (lane >> 4) * 8;
#pragma unroll
  for (int t = 0; t < 2; ++t) {
#pragma unroll
    for (int v = 0; v < 8; ++v) {
      const int s = rowBase + w * 32 + t * 16 + rh + v;
      if (MODE == 3) {
#pragma unroll
        for (int c = 0; c < 4; ++c)
          outf[(size_t)s * N + colBase + c * 16 + lc] = acc[t][c][v];
      } else if (MODE == 2) {
        const int kvh = colBase >> 6;
#pragma unroll
        for (int c = 0; c < 4; ++c) {
          const int d = c * 16 + lc;
          outb[(size_t)(kvh * HD + d) * S_LEN + s] = f2bf(acc[t][c][v]);
        }
      } else {  // RoPE: dims d and d+32 pair within this 64-wide head tile
#pragma unroll
        for (int c = 0; c < 2; ++c) {
          const int d = c * 16 + lc;  // 0..31
          const float x1 = acc[t][c][v], x2 = acc[t][c + 2][v];
          // inv_freq[d] = 10000^(-d/32) = exp(-d * ln(10000)/32)
          const float ang = (float)s * __expf(-0.28782313365f * (float)d);
          float sn, cs;
          __sincosf(ang, &sn, &cs);
          float o1 = x1 * cs - x2 * sn;
          float o2 = x2 * cs + x1 * sn;
          if (MODE == 0) {  // fold 1/sqrt(HD) into Q
            o1 *= 0.125f;
            o2 *= 0.125f;
            outb[(size_t)s * H_DIM + colBase + d]      = f2bf(o1);
            outb[(size_t)s * H_DIM + colBase + d + 32] = f2bf(o2);
          } else {
            const int kvh = colBase >> 6;
            outb[((size_t)kvh * S_LEN + s) * HD + d]      = f2bf(o1);
            outb[((size_t)kvh * S_LEN + s) * HD + d + 32] = f2bf(o2);
          }
        }
      }
    }
  }
}

// ---------------------------------------------------------------------------
// Flash attention: grid (S/128 query blocks, NH heads), 128 threads (4 waves).
// Wave w owns 32 queries (two 16-row WMMA tiles); every staged K/V fragment
// feeds BOTH tiles' WMMAs. 64-key K/V tiles double-buffered in LDS with async
// direct-to-LDS so the next block's DMA overlaps QK^T/softmax/PV.
// ---------------------------------------------------------------------------
__global__ __launch_bounds__(128) void flash_attn(
    const unsigned short* __restrict__ Qb,   // [S][H] bf16 (RoPE + scale)
    const unsigned short* __restrict__ Kb,   // [KVH][S][HD] bf16 (RoPE)
    const unsigned short* __restrict__ Vt,   // [KVH][HD][S] bf16
    unsigned short* __restrict__ Ob) {       // [S][H] bf16
  // buffer b (16KB): K [key][hd] @ byte b*16384, V [hd][key] @ +8192;
  // P staging: 4 waves x [32][64] @ byte 32768 (16KB). Total 48KB.
  __shared__ unsigned short smem[2 * 8192 + 8192];
  unsigned short* Ps = smem + 16384;

  const int tid = threadIdx.x, lane = tid & 31, w = tid >> 5;
  const int qb0 = blockIdx.x * 128;
  const int h = blockIdx.y;
  const int kvh = h & (KVH - 1);     // jnp.tile -> head n uses kv head n%8
  const int q0w = qb0 + w * 32;
  const int lc = lane & 15, rh = (lane >> 4) * 8;

  // Q fragments straight from global (A layout, HD=64 -> two K-chunks)
  v16bf qa[2][2];
#pragma unroll
  for (int t = 0; t < 2; ++t) {
    const unsigned short* qp = Qb + (size_t)(q0w + t * 16) * H_DIM + h * HD;
    qa[t][0] = load_frag_a(qp, H_DIM, 0, lane);
    qa[t][1] = load_frag_a(qp, H_DIM, 32, lane);
  }

  v8f o[2][4];
  float m[2][8], l[2][8];
#pragma unroll
  for (int t = 0; t < 2; ++t) {
#pragma unroll
    for (int c = 0; c < 4; ++c) o[t][c] = (v8f)(0.0f);
#pragma unroll
    for (int v = 0; v < 8; ++v) { m[t][v] = -__builtin_inff(); l[t][v] = 0.0f; }
  }

  const unsigned short* Kg = Kb + (size_t)kvh * S_LEN * HD;
  const unsigned short* Vg = Vt + (size_t)kvh * HD * S_LEN;
  const int nkb = qb0 / 64 + 2;      // causal: keys <= qb0+127

  auto stage = [&](int kb, int buf) {
    // K: contiguous 8KB; V^T: 64 rows x 128B with row stride 2*S bytes
    const unsigned lbase = (unsigned)(buf * 16384);
#pragma unroll
    for (int j = 0; j < 4; ++j) {
      const int id = tid + j * 128;
      async_ld_b128(lbase + (unsigned)(id * 16), Kg,
                    (unsigned)(kb * 8192 + id * 16));
      const int row = id >> 3, part = (id & 7) * 16;
      async_ld_b128(lbase + (unsigned)(8192 + row * 128 + part), Vg,
                    (unsigned)(row * (S_LEN * 2) + kb * 128 + part));
    }
  };

  stage(0, 0);
  wait_async();
  __syncthreads();

  for (int kb = 0; kb < nkb; ++kb) {
    const int cur = kb & 1;
    if (kb + 1 < nkb) stage(kb + 1, cur ^ 1);  // DMA overlaps compute below

    if (kb * 64 <= q0w + 31) {  // wave-uniform: EXEC all-ones at every WMMA
      const unsigned short* Ks = smem + cur * 8192;
      const unsigned short* Vs = Ks + 4096;

      // ---- scores S = Q K^T; each bk fragment feeds both query tiles ----
      v8f sc[2][4];
#pragma unroll
      for (int t = 0; t < 2; ++t)
#pragma unroll
        for (int g = 0; g < 4; ++g) sc[t][g] = (v8f)(0.0f);
#pragma unroll
      for (int kc = 0; kc < 2; ++kc)
#pragma unroll
        for (int g = 0; g < 4; ++g) {
          v16bf bk = load_frag_b(Ks, g * 16, 64, kc * 32, lane);
          sc[0][g] = wmma_bf16(qa[0][kc], bk, sc[0][g]);
          sc[1][g] = wmma_bf16(qa[1][kc], bk, sc[1][g]);
        }

      // ---- causal mask + online softmax (fp32) + P staging ----
      unsigned short* pw = Ps + w * 2048;
#pragma unroll
      for (int t = 0; t < 2; ++t) {
#pragma unroll
        for (int v = 0; v < 8; ++v) {
          const int qi = q0w + t * 16 + rh + v;
          float rmax = -__builtin_inff();
#pragma unroll
          for (int g = 0; g < 4; ++g) {
            const int key = kb * 64 + g * 16 + lc;
            float sv = (key > qi) ? -1.0e9f : sc[t][g][v];
            sc[t][g][v] = sv;
            rmax = fmaxf(rmax, sv);
          }
#pragma unroll
          for (int mk = 1; mk < 16; mk <<= 1)
            rmax = fmaxf(rmax, __shfl_xor(rmax, mk, 16));
          const float mnew = fmaxf(m[t][v], rmax);
          const float a0 = expf(m[t][v] - mnew);
          float rs = 0.0f;
#pragma unroll
          for (int g = 0; g < 4; ++g) {
            const float p = __expf(sc[t][g][v] - mnew);
            sc[t][g][v] = p;
            rs += p;
          }
#pragma unroll
          for (int mk = 1; mk < 16; mk <<= 1) rs += __shfl_xor(rs, mk, 16);
          l[t][v] = l[t][v] * a0 + rs;
          m[t][v] = mnew;
#pragma unroll
          for (int c = 0; c < 4; ++c) o[t][c][v] *= a0;
          // re-fragment P via LDS (C/D layout -> A layout), rows t*16+...
#pragma unroll
          for (int g = 0; g < 4; ++g)
            pw[(t * 16 + rh + v) * 64 + g * 16 + lc] = f2bf(sc[t][g][v]);
        }
      }
      asm volatile("s_wait_dscnt 0x0" ::: "memory");  // DS in-order per wave

      // ---- O += P V; each bv fragment feeds both query tiles ----
#pragma unroll
      for (int kc = 0; kc < 2; ++kc) {
        v16bf pa0 = load_frag_a(pw, 64, kc * 32, lane);
        v16bf pa1 = load_frag_a(pw + 16 * 64, 64, kc * 32, lane);
#pragma unroll
        for (int c = 0; c < 4; ++c) {
          v16bf bv = load_frag_b(Vs, c * 16, 64, kc * 32, lane);
          o[0][c] = wmma_bf16(pa0, bv, o[0][c]);
          o[1][c] = wmma_bf16(pa1, bv, o[1][c]);
        }
      }
    }

    wait_async();     // next K/V buffer's DMA done (hidden behind compute)
    __syncthreads();  // all waves done with cur buffer, deposits visible
  }

  // ---- normalize and write [S][H] bf16 ----
#pragma unroll
  for (int t = 0; t < 2; ++t)
#pragma unroll
    for (int v = 0; v < 8; ++v) {
      const float inv = 1.0f / l[t][v];
      const int qi = q0w + t * 16 + rh + v;
#pragma unroll
      for (int c = 0; c < 4; ++c)
        Ob[(size_t)qi * H_DIM + h * HD + c * 16 + lc] = f2bf(o[t][c][v] * inv);
    }
}

// ---------------------------------------------------------------------------
extern "C" void kernel_launch(void* const* d_in, const int* in_sizes, int n_in,
                              void* d_out, int out_size, void* d_ws,
                              size_t ws_size, hipStream_t stream) {
  (void)in_sizes; (void)n_in; (void)out_size; (void)ws_size;
  const float* hs = (const float*)d_in[0];
  const float* Wq = (const float*)d_in[1];
  const float* Wk = (const float*)d_in[2];
  const float* Wv = (const float*)d_in[3];
  const float* Wo = (const float*)d_in[4];
  float* out = (float*)d_out;

  // workspace layout (bf16 buffers), total ~48 MB
  unsigned short* hsb = (unsigned short*)d_ws;
  unsigned short* wqb = hsb + (size_t)S_LEN * H_DIM;
  unsigned short* wkb = wqb + (size_t)H_DIM * H_DIM;
  unsigned short* wvb = wkb + (size_t)H_DIM * (KVH * HD);
  unsigned short* wob = wvb + (size_t)H_DIM * (KVH * HD);
  unsigned short* qb  = wob + (size_t)H_DIM * H_DIM;
  unsigned short* kbf = qb  + (size_t)S_LEN * H_DIM;
  unsigned short* vtb = kbf + (size_t)KVH * S_LEN * HD;
  unsigned short* atb = vtb + (size_t)KVH * HD * S_LEN;

  auto cvt = [&](const float* src, unsigned short* dst, size_t n) {
    int n4 = (int)(n / 4);
    cvt_bf16_v4<<<(n4 + 255) / 256, 256, 0, stream>>>(src, dst, n4);
  };
  cvt(hs, hsb, (size_t)S_LEN * H_DIM);
  cvt(Wq, wqb, (size_t)H_DIM * H_DIM);
  cvt(Wk, wkb, (size_t)H_DIM * KVH * HD);
  cvt(Wv, wvb, (size_t)H_DIM * KVH * HD);
  cvt(Wo, wob, (size_t)H_DIM * H_DIM);

  // projections (+RoPE/scale fused)
  gemm_wmma<0><<<dim3(H_DIM / 64, S_LEN / 128), 128, 0, stream>>>(
      hsb, wqb, qb, nullptr, H_DIM, H_DIM);
  gemm_wmma<1><<<dim3((KVH * HD) / 64, S_LEN / 128), 128, 0, stream>>>(
      hsb, wkb, kbf, nullptr, KVH * HD, H_DIM);
  gemm_wmma<2><<<dim3((KVH * HD) / 64, S_LEN / 128), 128, 0, stream>>>(
      hsb, wvb, vtb, nullptr, KVH * HD, H_DIM);

  // attention
  flash_attn<<<dim3(S_LEN / 128, NH), 128, 0, stream>>>(qb, kbf, vtb, atb);

  // output projection -> fp32
  gemm_wmma<3><<<dim3(H_DIM / 64, S_LEN / 128), 128, 0, stream>>>(
      atb, wob, nullptr, out, H_DIM, H_DIM);
}
